// VariableAttentionLayer_19207093747765
// MI455X (gfx1250) — matrix operand order repaired and checked by reading, not compiled, not run on hardware
//
#include <hip/hip_runtime.h>
#include <hip/hip_bf16.h>

// ---------------------------------------------------------------------------
// CDNA5 (gfx1250) implementation of VariableAttentionLayer forward.
// All matmuls via v_wmma_f32_16x16x32_bf16 (wave32). Every GEMM epilogue
// (bias/LN/GELU/residual/row-dot) is fused into the GEMM block (32x512 block
// tile = full rows), eliminating all f32 intermediate round trips.
// ---------------------------------------------------------------------------

typedef __attribute__((ext_vector_type(16))) __bf16 v16bf;
typedef __attribute__((ext_vector_type(8)))  float  v8f;

union FragBF {              // 16 bf16 elements = 32 bytes = 2 x uint4
    v16bf v;
    uint4 u[2];
    unsigned short e[16];
};

__device__ __forceinline__ v8f wmma_bf16(v16bf a, v16bf b, v8f c) {
    return __builtin_amdgcn_wmma_f32_16x16x32_bf16(false, a, false, b,
                                                   (short)0, c, false, false);
}

__device__ __forceinline__ unsigned short f32_to_bf16u(float f) {
    unsigned int u = __float_as_uint(f);
    unsigned int r = u + 0x7FFFu + ((u >> 16) & 1u);   // round to nearest even
    return (unsigned short)(r >> 16);
}

__device__ __forceinline__ float gelu_exact(float v) {
    return 0.5f * v * (1.0f + erff(v * 0.70710678118654752440f));
}

__device__ __forceinline__ float wave_sum32(float v) {
#pragma unroll
    for (int m = 1; m < 32; m <<= 1) v += __shfl_xor(v, m, 32);
    return v;
}
__device__ __forceinline__ float half_sum16(float v) {
#pragma unroll
    for (int m = 1; m < 16; m <<= 1) v += __shfl_xor(v, m, 32);
    return v;
}

// ---------------------------------------------------------------------------
// f32 -> bf16 convert (row-major [K,N] weights need no transpose: the CDNA5
// B-fragment is lane = N column, elements = 16 K rows read at stride N).
// ---------------------------------------------------------------------------
__global__ void k_f32_to_bf16(const float* __restrict__ src,
                              unsigned short* __restrict__ dst, int n) {
    int i = blockIdx.x * blockDim.x + threadIdx.x;
    if (i < n) dst[i] = f32_to_bf16u(src[i]);
}

// ---------------------------------------------------------------------------
// Fused WMMA GEMM: Y = A[M,512](bf16) * B[512,512](bf16) + bias, then a
// row-complete epilogue. Block tile 32 rows x 512 cols (8 waves x 64 cols),
// so every output row is entirely inside one block:
//   MODE 0 (LN):        out_bf = LN(Y; g,beta) * outScale
//   MODE 1 (GELU+RES+LN): y = LN(gelu(Y)+xres); out_f = y, out_bf = bf16(y)
//   MODE 2 (GELU+DOT):  out_f[row] = gelu(Y[row,:]) . w2 + b2
// K fully unrolled + double-buffered fragments (immediate-offset b128 loads).
// ---------------------------------------------------------------------------
enum { EPI_LN = 0, EPI_GELU_RES_LN = 1, EPI_GELU_DOT = 2 };

template <int MODE>
__global__ __launch_bounds__(256) void k_gemm_fused(
    const unsigned short* __restrict__ A,
    const unsigned short* __restrict__ Bw,
    const float* __restrict__ bias,
    const float* __restrict__ g,       // LN gamma     (modes 0,1)
    const float* __restrict__ beta,    // LN beta      (modes 0,1)
    const float* __restrict__ xres,    // residual     (mode 1)
    const float* __restrict__ w2,      // dot vector   (mode 2)
    const float* __restrict__ b2,      // dot bias     (mode 2)
    unsigned short* __restrict__ outb, // bf16 output  (modes 0,1)
    float* __restrict__ outf,          // f32 output (mode 1) / logits (mode 2)
    float outScale) {
    constexpr int N = 512, K = 512;
    const int lane = threadIdx.x & 31;
    const int wave = threadIdx.x >> 5;
    const int lo = lane & 15, hi = lane >> 4;
    const int m0 = blockIdx.x * 32;
    const int n0 = wave * 64;
    const int kb = hi * 8;

    v8f acc[2][4];
#pragma unroll
    for (int i = 0; i < 2; ++i)
#pragma unroll
        for (int t = 0; t < 4; ++t)
#pragma unroll
            for (int r = 0; r < 8; ++r) acc[i][t][r] = 0.0f;

    const unsigned short* ap0 = A + (size_t)(m0 + lo) * K + kb;
    const unsigned short* ap1 = ap0 + (size_t)16 * K;
    const unsigned short* bp  = Bw + (size_t)kb * N + n0 + lo;

    FragBF aX[2], bX[4], aY[2], bY[4];
    auto ldA = [&](FragBF* af, int k0) {
        af[0].u[0] = *(const uint4*)(const void*)(ap0 + k0);
        af[0].u[1] = *(const uint4*)(const void*)(ap0 + k0 + 16);
        af[1].u[0] = *(const uint4*)(const void*)(ap1 + k0);
        af[1].u[1] = *(const uint4*)(const void*)(ap1 + k0 + 16);
    };
    auto ldB = [&](FragBF* bf, int k0) {
        const unsigned short* p = bp + (size_t)k0 * N;
#pragma unroll
        for (int t = 0; t < 4; ++t) {
#pragma unroll
            for (int s = 0; s < 8; ++s) {
                bf[t].e[s]     = p[(size_t)s * N + t * 16];
                bf[t].e[8 + s] = p[(size_t)(16 + s) * N + t * 16];
            }
        }
    };
    auto mul = [&](FragBF* af, FragBF* bf) {
#pragma unroll
        for (int t = 0; t < 4; ++t) {
            acc[0][t] = wmma_bf16(af[0].v, bf[t].v, acc[0][t]);
            acc[1][t] = wmma_bf16(af[1].v, bf[t].v, acc[1][t]);
        }
    };

    ldA(aX, 0); ldB(bX, 0);
#pragma unroll
    for (int k0 = 0; k0 < K; k0 += 64) {
        ldA(aY, k0 + 32); ldB(bY, k0 + 32);
        mul(aX, bX);
        if (k0 + 64 < K) { ldA(aX, k0 + 64); ldB(bX, k0 + 64); }
        mul(aY, bY);
    }

    // ---- epilogue: bias + pointwise transform (in place) ----
#pragma unroll
    for (int i = 0; i < 2; ++i)
#pragma unroll
        for (int t = 0; t < 4; ++t) {
            const int col = n0 + t * 16 + lo;
            const float bv = bias[col];
#pragma unroll
            for (int r = 0; r < 8; ++r) {
                float v = acc[i][t][r] + bv;
                if constexpr (MODE == EPI_GELU_RES_LN) {
                    const int row = m0 + i * 16 + r + 8 * hi;
                    v = gelu_exact(v) + xres[(size_t)row * N + col];
                } else if constexpr (MODE == EPI_GELU_DOT) {
                    v = gelu_exact(v);
                }
                acc[i][t][r] = v;
            }
        }

    __shared__ float red_s[32][8];
    __shared__ float red_q[32][8];
    __shared__ float stat_m[32], stat_i[32];

    if constexpr (MODE == EPI_GELU_DOT) {
        // per-row dot with w2, reduced across cols: wave half -> LDS -> block
#pragma unroll
        for (int i = 0; i < 2; ++i)
#pragma unroll
            for (int r = 0; r < 8; ++r) {
                float s = 0.f;
#pragma unroll
                for (int t = 0; t < 4; ++t)
                    s += acc[i][t][r] * w2[n0 + t * 16 + lo];
                s = half_sum16(s);
                if (lo == 0) red_s[i * 16 + r + 8 * hi][wave] = s;
            }
        __syncthreads();
        if (threadIdx.x < 32) {
            float s = 0.f;
#pragma unroll
            for (int w = 0; w < 8; ++w) s += red_s[threadIdx.x][w];
            outf[m0 + threadIdx.x] = s + b2[0];
        }
        return;
    } else {
        // LayerNorm over the full 512-wide row
#pragma unroll
        for (int i = 0; i < 2; ++i)
#pragma unroll
            for (int r = 0; r < 8; ++r) {
                float s = 0.f, q = 0.f;
#pragma unroll
                for (int t = 0; t < 4; ++t) {
                    const float v = acc[i][t][r];
                    s += v; q += v * v;
                }
                s = half_sum16(s); q = half_sum16(q);
                if (lo == 0) {
                    const int row = i * 16 + r + 8 * hi;
                    red_s[row][wave] = s;
                    red_q[row][wave] = q;
                }
            }
        __syncthreads();
        if (threadIdx.x < 32) {
            float s = 0.f, q = 0.f;
#pragma unroll
            for (int w = 0; w < 8; ++w) { s += red_s[threadIdx.x][w]; q += red_q[threadIdx.x][w]; }
            const float mean = s * (1.0f / N);
            const float var = q * (1.0f / N) - mean * mean;
            stat_m[threadIdx.x] = mean;
            stat_i[threadIdx.x] = rsqrtf(var + 1e-5f);
        }
        __syncthreads();
#pragma unroll
        for (int i = 0; i < 2; ++i)
#pragma unroll
            for (int r = 0; r < 8; ++r) {
                const int lrow = i * 16 + r + 8 * hi;
                const int row = m0 + lrow;
                const float mean = stat_m[lrow];
                const float inv = stat_i[lrow];
#pragma unroll
                for (int t = 0; t < 4; ++t) {
                    const int col = n0 + t * 16 + lo;
                    const float y = (acc[i][t][r] - mean) * inv * g[col] + beta[col];
                    if constexpr (MODE == EPI_LN) {
                        outb[(size_t)row * N + col] = f32_to_bf16u(y * outScale);
                    } else {
                        outf[(size_t)row * N + col] = y;
                        outb[(size_t)row * N + col] = f32_to_bf16u(y);
                    }
                }
            }
    }
}

// ---------------------------------------------------------------------------
// Flash attention, one wave per (b, h, 16-query block), Dh = 64.
// Computes S^T = K * Q^T so that:
//   - K tiles load as contiguous A-fragments (no in-loop transpose),
//   - Q^T stages through LDS exactly once (loop-invariant B-fragments),
//   - S^T's C-layout (lane=query, reg=key) IS P's A-layout (lane=query,
//     elem=key): probabilities convert to bf16 fully in-register.
// Q buffer is pre-scaled by 1/sqrt(Dh).
// ---------------------------------------------------------------------------
__global__ __launch_bounds__(256) void k_attn(
    const unsigned short* __restrict__ qb,
    const unsigned short* __restrict__ kbuf,
    const unsigned short* __restrict__ vbuf,
    unsigned short* __restrict__ ob) {
    constexpr int C = 512, D = 512, Dh = 64;
    const int lane = threadIdx.x & 31;
    const int wave = threadIdx.x >> 5;
    const int lo = lane & 15, hi = lane >> 4;
    const int unit = blockIdx.x * 8 + wave;      // B*H*(C/16) = 8192 units
    const int b = unit >> 8;
    const int rem = unit & 255;
    const int h = rem >> 5;
    const int m0 = (rem & 31) << 4;
    const int kb = hi * 8;

    __shared__ alignas(16) unsigned short qt_s[8][64 * 16];  // [query][d]
    unsigned short* qt = qt_s[wave];

    // --- one-time Q^T staging: lane handles query lo, d-half hi*32 ---
    {
        const unsigned short* qrow =
            qb + ((size_t)(b * C + m0 + lo)) * D + h * Dh + hi * 32;
#pragma unroll
        for (int c = 0; c < 4; ++c) {
            uint4 d4 = *(const uint4*)(const void*)(qrow + c * 8);
            const unsigned short* e = (const unsigned short*)&d4;
#pragma unroll
            for (int j = 0; j < 8; ++j) qt[lo * 64 + hi * 32 + c * 8 + j] = e[j];
        }
    }
    // --- loop-invariant Q^T B-fragments: lane = query column, elems = d ---
    FragBF qbf[2];
#pragma unroll
    for (int c = 0; c < 2; ++c) {
        const unsigned short* p = qt + lo * 64 + c * 32 + kb;
        qbf[c].u[0] = *(const uint4*)(const void*)(p);
        qbf[c].u[1] = *(const uint4*)(const void*)(p + 16);
    }

    float m_run = -1e30f, l_run = 0.f;           // per query (= lane lo)
    v8f o[4];
#pragma unroll
    for (int t = 0; t < 4; ++t)
#pragma unroll
        for (int r = 0; r < 8; ++r) o[t][r] = 0.f;

    for (int s0 = 0; s0 < C; s0 += 32) {
        const unsigned short* kbase = kbuf + ((size_t)(b * C + s0)) * D + h * Dh;
        const unsigned short* vbase = vbuf + ((size_t)(b * C + s0)) * D + h * Dh;
        if (s0 + 32 < C) {                        // next tile -> global_prefetch
            __builtin_prefetch(kbase + (size_t)32 * D + (size_t)lane * D, 0, 3);
            __builtin_prefetch(vbase + (size_t)32 * D + (size_t)lane * D, 0, 3);
        }
        // --- K A-fragments (rows = keys), contiguous global loads ---
        FragBF kf[2][2];                          // [key subtile][d chunk]
#pragma unroll
        for (int u = 0; u < 2; ++u) {
            const unsigned short* krow = kbase + (size_t)(u * 16 + lo) * D;
#pragma unroll
            for (int c = 0; c < 2; ++c) {
                kf[u][c].u[0] = *(const uint4*)(const void*)(krow + c * 32 + kb);
                kf[u][c].u[1] = *(const uint4*)(const void*)(krow + c * 32 + 16 + kb);
            }
        }
        // --- S^T tiles: reg r = key r+8*hi (+16*u), lane lo = query ---
        v8f sc[2];
#pragma unroll
        for (int u = 0; u < 2; ++u) {
#pragma unroll
            for (int r = 0; r < 8; ++r) sc[u][r] = 0.f;
            sc[u] = wmma_bf16(kf[u][0].v, qbf[0].v, sc[u]);
            sc[u] = wmma_bf16(kf[u][1].v, qbf[1].v, sc[u]);
        }
        // --- online softmax over keys (in-register + one shfl) ---
        float mx = sc[0][0];
#pragma unroll
        for (int u = 0; u < 2; ++u)
#pragma unroll
            for (int r = 0; r < 8; ++r) mx = fmaxf(mx, sc[u][r]);
        mx = fmaxf(mx, __shfl_xor(mx, 16, 32));
        const float mnew = fmaxf(m_run, mx);
        const float alpha = __expf(m_run - mnew);
        float p[2][8];
        float rs = 0.f;
#pragma unroll
        for (int u = 0; u < 2; ++u)
#pragma unroll
            for (int r = 0; r < 8; ++r) {
                p[u][r] = __expf(sc[u][r] - mnew);
                rs += p[u][r];
            }
        rs += __shfl_xor(rs, 16, 32);
        l_run = l_run * alpha + rs;
        m_run = mnew;
        // --- redistribute alpha from query-in-lane to query-in-register ---
        float alpha_row[8];
#pragma unroll
        for (int r = 0; r < 8; ++r) alpha_row[r] = __shfl(alpha, r + 8 * hi, 32);
#pragma unroll
        for (int t = 0; t < 4; ++t)
#pragma unroll
            for (int r = 0; r < 8; ++r) o[t][r] *= alpha_row[r];
        // --- P: S^T C-layout == P A-layout; pack bf16 fully in-register ---
        FragBF pf;
#pragma unroll
        for (int e = 0; e < 8; ++e) {
            pf.e[e]     = f32_to_bf16u(p[0][e]);
            pf.e[8 + e] = f32_to_bf16u(p[1][e]);
        }
        // --- V B-fragments straight from global (lane = d column, elems = keys) ---
        const unsigned short* vrow = vbase + (size_t)kb * D + lo;
#pragma unroll
        for (int t = 0; t < 4; ++t) {
            FragBF vf;
#pragma unroll
            for (int s = 0; s < 8; ++s) {
                vf.e[s]     = vrow[(size_t)s * D + t * 16];
                vf.e[8 + s] = vrow[(size_t)(16 + s) * D + t * 16];
            }
            o[t] = wmma_bf16(pf.v, vf.v, o[t]);
        }
    }

    const float inv = 1.0f / l_run;
    float inv_row[8];
#pragma unroll
    for (int r = 0; r < 8; ++r) inv_row[r] = __shfl(inv, r + 8 * hi, 32);
#pragma unroll
    for (int r = 0; r < 8; ++r) {
        const size_t base = ((size_t)(b * C + m0 + r + 8 * hi)) * D + h * Dh;
#pragma unroll
        for (int t = 0; t < 4; ++t)
            ob[base + t * 16 + lo] = f32_to_bf16u(o[t][r] * inv_row[r]);
    }
}

// ---------------------------------------------------------------------------
// fi = softmax over C=512 per batch (wave per batch).
// ---------------------------------------------------------------------------
__global__ __launch_bounds__(256) void k_softmax_fi(
    const float* __restrict__ wlog, float* __restrict__ fi) {
    constexpr int C = 512;
    const int lane = threadIdx.x & 31;
    const int wave = threadIdx.x >> 5;
    const int bidx = blockIdx.x * 8 + wave;
    const float* p = wlog + (size_t)bidx * C;
    float vals[16];
    float mx = -1e30f;
#pragma unroll
    for (int i = 0; i < 16; ++i) {
        vals[i] = p[lane + i * 32];
        mx = fmaxf(mx, vals[i]);
    }
#pragma unroll
    for (int m = 1; m < 32; m <<= 1) mx = fmaxf(mx, __shfl_xor(mx, m, 32));
    float s = 0.f;
#pragma unroll
    for (int i = 0; i < 16; ++i) { vals[i] = __expf(vals[i] - mx); s += vals[i]; }
    s = wave_sum32(s);
    const float inv = 1.0f / s;
#pragma unroll
    for (int i = 0; i < 16; ++i)
        fi[(size_t)bidx * C + lane + i * 32] = vals[i] * inv;
}

// ---------------------------------------------------------------------------
// Host-side orchestration.
// ---------------------------------------------------------------------------
extern "C" void kernel_launch(void* const* d_in, const int* in_sizes, int n_in,
                              void* d_out, int out_size, void* d_ws, size_t ws_size,
                              hipStream_t stream) {
    (void)in_sizes; (void)n_in; (void)out_size; (void)ws_size;
    constexpr int B = 32, C = 512, Dm = 512;
    constexpr int BC = B * C;                 // 16384 rows

    const float* x    = (const float*)d_in[0];
    const float* Wq   = (const float*)d_in[1];
    const float* bq   = (const float*)d_in[2];
    const float* Wk   = (const float*)d_in[3];
    const float* bk   = (const float*)d_in[4];
    const float* Wv   = (const float*)d_in[5];
    const float* bv   = (const float*)d_in[6];
    const float* gq   = (const float*)d_in[7];
    const float* Bq   = (const float*)d_in[8];
    const float* gk   = (const float*)d_in[9];
    const float* Bk   = (const float*)d_in[10];
    const float* gv   = (const float*)d_in[11];
    const float* Bv   = (const float*)d_in[12];
    const float* Wo   = (const float*)d_in[13];
    const float* bo   = (const float*)d_in[14];
    const float* g_ln = (const float*)d_in[15];
    const float* b_ln = (const float*)d_in[16];
    const float* W1   = (const float*)d_in[17];
    const float* b1   = (const float*)d_in[18];
    const float* W2   = (const float*)d_in[19];
    const float* b2   = (const float*)d_in[20];

    float* x2_out = (float*)d_out;
    float* fi_out = (float*)d_out + (size_t)BC * Dm;

    // ---- workspace carve-out ----
    char* w = (char*)d_ws;
    auto carve = [&](size_t bytes) -> char* {
        char* p = w;
        w += (bytes + 255) & ~(size_t)255;
        return p;
    };
    const size_t act_b16 = (size_t)BC * Dm * sizeof(unsigned short);  // 16 MiB
    const size_t wgt_b16 = (size_t)Dm * Dm * sizeof(unsigned short);  // 512 KiB
    unsigned short* xb    = (unsigned short*)carve(act_b16);
    unsigned short* qb    = (unsigned short*)carve(act_b16);
    unsigned short* kb_   = (unsigned short*)carve(act_b16);
    unsigned short* vb_   = (unsigned short*)carve(act_b16);
    unsigned short* attnb = (unsigned short*)carve(act_b16);
    unsigned short* x2b   = (unsigned short*)carve(act_b16);
    unsigned short* wqb   = (unsigned short*)carve(wgt_b16);
    unsigned short* wkb   = (unsigned short*)carve(wgt_b16);
    unsigned short* wvb   = (unsigned short*)carve(wgt_b16);
    unsigned short* wob   = (unsigned short*)carve(wgt_b16);
    unsigned short* w1b   = (unsigned short*)carve(wgt_b16);
    float* wlog = (float*)carve((size_t)BC * sizeof(float));

    // ---- precision conversion ----
    k_f32_to_bf16<<<(BC * Dm + 255) / 256, 256, 0, stream>>>(x, xb, BC * Dm);
    k_f32_to_bf16<<<(Dm * Dm + 255) / 256, 256, 0, stream>>>(Wq, wqb, Dm * Dm);
    k_f32_to_bf16<<<(Dm * Dm + 255) / 256, 256, 0, stream>>>(Wk, wkb, Dm * Dm);
    k_f32_to_bf16<<<(Dm * Dm + 255) / 256, 256, 0, stream>>>(Wv, wvb, Dm * Dm);
    k_f32_to_bf16<<<(Dm * Dm + 255) / 256, 256, 0, stream>>>(Wo, wob, Dm * Dm);
    k_f32_to_bf16<<<(Dm * Dm + 255) / 256, 256, 0, stream>>>(W1, w1b, Dm * Dm);

    const int gblocks = BC / 32;              // 512 blocks, 32 rows each
    // ---- Q/K/V projections with fused LayerNorm (Q pre-scaled 1/sqrt(Dh)) ----
    k_gemm_fused<EPI_LN><<<gblocks, 256, 0, stream>>>(
        xb, wqb, bq, gq, Bq, nullptr, nullptr, nullptr, qb, nullptr, 0.125f);
    k_gemm_fused<EPI_LN><<<gblocks, 256, 0, stream>>>(
        xb, wkb, bk, gk, Bk, nullptr, nullptr, nullptr, kb_, nullptr, 1.0f);
    k_gemm_fused<EPI_LN><<<gblocks, 256, 0, stream>>>(
        xb, wvb, bv, gv, Bv, nullptr, nullptr, nullptr, vb_, nullptr, 1.0f);

    // ---- attention: 8192 wave-units / 8 waves per block ----
    k_attn<<<1024, 256, 0, stream>>>(qb, kb_, vb_, attnb);

    // ---- output projection with fused GELU + residual + LN ----
    k_gemm_fused<EPI_GELU_RES_LN><<<gblocks, 256, 0, stream>>>(
        attnb, wob, bo, g_ln, b_ln, x, nullptr, nullptr, x2b, x2_out, 1.0f);

    // ---- feature-importance MLP (W1+GELU+dot W2 fused) + softmax over C ----
    k_gemm_fused<EPI_GELU_DOT><<<gblocks, 256, 0, stream>>>(
        x2b, w1b, b1, nullptr, nullptr, nullptr, W2, b2, nullptr, wlog, 1.0f);
    k_softmax_fi<<<B / 8, 256, 0, stream>>>(wlog, fi_out);
}
